// mLSTMBlock_8744553415141
// MI455X (gfx1250) — compile-verified, matrix operand order checked by reading, not run on hardware
//
#include <hip/hip_runtime.h>
#include <hip/hip_bf16.h>
#include <math.h>

#define Bc 2
#define Sc 2048
#define Dc 1024
#define NHc 8
#define DHc 128

typedef __attribute__((ext_vector_type(16))) __bf16 v16bf;
typedef __attribute__((ext_vector_type(8)))  float  v8f;

__device__ __forceinline__ v8f wmma_bf16(v16bf a, v16bf b, v8f c) {
  return __builtin_amdgcn_wmma_f32_16x16x32_bf16(false, a, false, b, (short)0, c,
                                                 false, false);
}

// ---------------------------------------------------------------- converts
__global__ void __launch_bounds__(256) cvt_bf16_kernel(const float* __restrict__ src,
                                                       __bf16* __restrict__ dst, int n) {
  int i = blockIdx.x * 256 + threadIdx.x;
  if (i < n) dst[i] = (__bf16)src[i];
}

__global__ void __launch_bounds__(256) cvt_bf16_scale_kernel(const float* __restrict__ src,
                                                             __bf16* __restrict__ dst,
                                                             float s, int n) {
  int i = blockIdx.x * 256 + threadIdx.x;
  if (i < n) dst[i] = (__bf16)(src[i] * s);
}

// v [B*S, DH] f32  ->  vT [B, DH, S] bf16
__global__ void __launch_bounds__(256) transpose_v_kernel(const float* __restrict__ v,
                                                          __bf16* __restrict__ vT) {
  int i = blockIdx.x * 256 + threadIdx.x;
  if (i >= Bc * Sc * DHc) return;
  int d = i % DHc;
  int s = (i / DHc) % Sc;
  int b = i / (DHc * Sc);
  vT[(size_t)(b * DHc + d) * Sc + s] = (__bf16)v[i];
}

// ---------------------------------------------------------------- bf16 GEMM
// Out[m,n] = sum_k A[m,k] * W[n,k]   (A:[M,K], W:[N,K] both row-major bf16)
// block = 256 threads = 8 waves arranged 2(M) x 4(N); wave tile 64x64
// (4x4 WMMA accumulators, 16 WMMAs per 8 fragment loads per K-step).
__global__ void __launch_bounds__(256) gemm_bf16_kernel(const __bf16* __restrict__ A,
                                                        const __bf16* __restrict__ W,
                                                        float* __restrict__ Out,
                                                        int M, int N, int K) {
  const int tid  = threadIdx.x;
  const int wave = tid >> 5, lane = tid & 31;
  const int r    = lane & 15, half = lane >> 4;
  const int wm   = wave & 1,  wn   = wave >> 1;
  const int m0   = (blockIdx.x * 2 + wm) * 64;
  const int n0   = (blockIdx.y * 4 + wn) * 64;
  if (m0 >= M || n0 >= N) return;

  v8f z = {0.f, 0.f, 0.f, 0.f, 0.f, 0.f, 0.f, 0.f};
  v8f c[4][4];
#pragma unroll
  for (int i = 0; i < 4; i++)
#pragma unroll
    for (int j = 0; j < 4; j++) c[i][j] = z;

  const __bf16* ap = A + (size_t)(m0 + r) * K + half * 16;
  const __bf16* wp = W + (size_t)(n0 + r) * K + half * 16;

  for (int k0 = 0; k0 < K; k0 += 32) {
    v16bf a[4];
#pragma unroll
    for (int i = 0; i < 4; i++) a[i] = *(const v16bf*)(ap + (size_t)i * 16 * K + k0);
#pragma unroll
    for (int j = 0; j < 4; j++) {
      v16bf bj = *(const v16bf*)(wp + (size_t)j * 16 * K + k0);
#pragma unroll
      for (int i = 0; i < 4; i++) c[i][j] = wmma_bf16(a[i], bj, c[i][j]);
    }
  }
#pragma unroll
  for (int i = 0; i < 4; i++)
#pragma unroll
    for (int j = 0; j < 4; j++)
#pragma unroll
      for (int rr = 0; rr < 8; rr++)
        Out[(size_t)(m0 + i * 16 + half * 8 + rr) * N + n0 + j * 16 + r] = c[i][j][rr];
}

// ---------------------------------------------------------------- gates
// One wave per (b,s): i_pre = softcap(x.Wi+bi), f_pre = softcap(x.Wf+bf),
// store i_pre and log_sigmoid(f_pre), layout [B,NH,S].
__global__ void __launch_bounds__(256) gates_kernel(const float* __restrict__ x,
                                                    const float* __restrict__ Wi,
                                                    const float* __restrict__ biv,
                                                    const float* __restrict__ Wf,
                                                    const float* __restrict__ bfv,
                                                    float* __restrict__ ipre,
                                                    float* __restrict__ logfv) {
  const int wid  = blockIdx.x * 8 + (threadIdx.x >> 5);  // 0..B*S-1
  const int lane = threadIdx.x & 31;
  const int b    = wid >> 11;
  const int s    = wid & (Sc - 1);
  const float* xr = x + (size_t)(b * Sc + s) * Dc;
  float xv[32];
#pragma unroll
  for (int i = 0; i < 32; i++) xv[i] = xr[lane + i * 32];
  for (int h = 0; h < NHc; h++) {
    float di = 0.f, df = 0.f;
#pragma unroll
    for (int i = 0; i < 32; i++) {
      di += xv[i] * Wi[(size_t)h * Dc + lane + i * 32];
      df += xv[i] * Wf[(size_t)h * Dc + lane + i * 32];
    }
#pragma unroll
    for (int m = 1; m < 32; m <<= 1) {
      di += __shfl_xor(di, m, 32);
      df += __shfl_xor(df, m, 32);
    }
    if (lane == 0) {
      float ic = 15.f * tanhf((di + biv[h]) * (1.f / 15.f));
      float fc = 15.f * tanhf((df + bfv[h]) * (1.f / 15.f));
      float ls = fminf(fc, 0.f) - log1pf(__expf(-fabsf(fc)));
      ipre[(size_t)(b * NHc + h) * Sc + s]  = ic;
      logfv[(size_t)(b * NHc + h) * Sc + s] = ls;
    }
  }
}

// ---------------------------------------------------------------- scan
// Per (b,h): csum = inclusive cumsum(log_f); ucomb = i_pre - csum.
__global__ void __launch_bounds__(256) scan_kernel(const float* __restrict__ logfv,
                                                   const float* __restrict__ ipre,
                                                   float* __restrict__ csum,
                                                   float* __restrict__ ucomb) {
  const int bh = blockIdx.x;
  const float* src = logfv + (size_t)bh * Sc;
  const int t = threadIdx.x;
  float v[8];
  float run = 0.f;
#pragma unroll
  for (int j = 0; j < 8; j++) {
    run += src[t * 8 + j];
    v[j] = run;
  }
  __shared__ float tot[256];
  tot[t] = run;
  __syncthreads();
  for (int off = 1; off < 256; off <<= 1) {
    float add = (t >= off) ? tot[t - off] : 0.f;
    __syncthreads();
    tot[t] += add;
    __syncthreads();
  }
  float excl = tot[t] - run;
#pragma unroll
  for (int j = 0; j < 8; j++) {
    int idx  = t * 8 + j;
    float cv = excl + v[j];
    csum[(size_t)bh * Sc + idx]  = cv;
    ucomb[(size_t)bh * Sc + idx] = ipre[(size_t)bh * Sc + idx] - cv;
  }
}

// ---------------------------------------------------------------- flash mLSTM
// One wave per 16-row q strip of one (b,h). Streams 32-col blocks with
// online max/renorm. h_buf layout [B,S,NH,DH] f32.
__global__ void __launch_bounds__(256) mlstm_attn_kernel(const __bf16* __restrict__ qbf,
                                                         const __bf16* __restrict__ kbf,
                                                         const __bf16* __restrict__ vT,
                                                         const float* __restrict__ csum,
                                                         const float* __restrict__ ucomb,
                                                         float* __restrict__ hbuf) {
  __shared__ __bf16 lds[8][16 * 32];
  const int tid  = threadIdx.x;
  const int wave = tid >> 5, lane = tid & 31;
  const int r    = lane & 15, half = lane >> 4;
  const int gw   = blockIdx.x * 8 + wave;
  const int is   = gw & 127;
  const int h    = (gw >> 7) & 7;
  const int b    = gw >> 10;
  const int s0   = is << 4;
  const int bh   = b * NHc + h;

  v16bf aq[4];
  const __bf16* qrow = qbf + (size_t)(b * Sc + s0 + r) * Dc + h * DHc + half * 16;
#pragma unroll
  for (int kk = 0; kk < 4; kk++) aq[kk] = *(const v16bf*)(qrow + kk * 32);

  float cm[8];
#pragma unroll
  for (int rr = 0; rr < 8; rr++) cm[rr] = csum[(size_t)bh * Sc + s0 + rr + half * 8];

  v8f z = {0.f, 0.f, 0.f, 0.f, 0.f, 0.f, 0.f, 0.f};
  v8f acc[8];
  float m_run[8], b_run[8];
#pragma unroll
  for (int dn = 0; dn < 8; dn++) acc[dn] = z;
#pragma unroll
  for (int rr = 0; rr < 8; rr++) {
    m_run[rr] = -INFINITY;
    b_run[rr] = 0.f;
  }

  __bf16* lp = &lds[wave][0];
  const int nb = (s0 + 47) >> 5;  // causal: cover t in [0, s0+16)

  for (int jc = 0; jc < nb; jc++) {
    const int t0 = jc << 5;
    // ---- S = q . k^T for the two 16-col subtiles
    v8f sa0 = z, sa1 = z;
    const __bf16* k0p = kbf + (size_t)(b * Sc + t0 + r) * DHc + half * 16;
    const __bf16* k1p = k0p + (size_t)16 * DHc;
#pragma unroll
    for (int kk = 0; kk < 4; kk++) {
      v16bf bk = *(const v16bf*)(k0p + kk * 32);
      sa0 = wmma_bf16(aq[kk], bk, sa0);
    }
#pragma unroll
    for (int kk = 0; kk < 4; kk++) {
      v16bf bk = *(const v16bf*)(k1p + kk * 32);
      sa1 = wmma_bf16(aq[kk], bk, sa1);
    }
    // ---- decay logits + online max/renorm
    const int   tc0 = t0 + r, tc1 = t0 + 16 + r;
    const float u0  = ucomb[(size_t)bh * Sc + tc0];
    const float u1  = ucomb[(size_t)bh * Sc + tc1];
    float scale[8], mnew[8], d0v[8], d1v[8];
#pragma unroll
    for (int rr = 0; rr < 8; rr++) {
      const int m = s0 + rr + half * 8;
      float d0 = (tc0 <= m) ? (cm[rr] + u0) : -INFINITY;
      float d1 = (tc1 <= m) ? (cm[rr] + u1) : -INFINITY;
      d0v[rr] = d0;
      d1v[rr] = d1;
      float mx = fmaxf(d0, d1);
      mx = fmaxf(mx, __shfl_xor(mx, 1, 32));
      mx = fmaxf(mx, __shfl_xor(mx, 2, 32));
      mx = fmaxf(mx, __shfl_xor(mx, 4, 32));
      mx = fmaxf(mx, __shfl_xor(mx, 8, 32));
      float mn  = fmaxf(m_run[rr], mx);
      scale[rr] = __expf(m_run[rr] - mn);
      mnew[rr]  = mn;
      m_run[rr] = mn;
    }
    // ---- C = S * exp(dlog - m); denominator update; stash C tile in LDS (bf16)
#pragma unroll
    for (int rr = 0; rr < 8; rr++) {
      float c0 = sa0[rr] * __expf(d0v[rr] - mnew[rr]);
      float c1 = sa1[rr] * __expf(d1v[rr] - mnew[rr]);
      float rs = c0 + c1;
      rs += __shfl_xor(rs, 1, 32);
      rs += __shfl_xor(rs, 2, 32);
      rs += __shfl_xor(rs, 4, 32);
      rs += __shfl_xor(rs, 8, 32);
      b_run[rr] = b_run[rr] * scale[rr] + rs;
      lp[(rr + half * 8) * 32 + r]      = (__bf16)c0;
      lp[(rr + half * 8) * 32 + 16 + r] = (__bf16)c1;
    }
    // ---- rescale running numerator
#pragma unroll
    for (int dn = 0; dn < 8; dn++)
#pragma unroll
      for (int rr = 0; rr < 8; rr++) acc[dn][rr] *= scale[rr];
    // ---- accumulate C (16x32) x V (32x128) via WMMA
    v16bf ac = *(const v16bf*)(lp + r * 32 + half * 16);
    const __bf16* vbase = vT + (size_t)(b * DHc) * Sc + t0 + half * 16;
#pragma unroll
    for (int dn = 0; dn < 8; dn++) {
      v16bf bv = *(const v16bf*)(vbase + (size_t)(dn * 16 + r) * Sc);
      acc[dn]  = wmma_bf16(ac, bv, acc[dn]);
    }
  }
  // ---- finalize: n = max(|b|, exp(-m)); h = acc / (n + eps)
#pragma unroll
  for (int rr = 0; rr < 8; rr++) {
    float n   = fmaxf(fabsf(b_run[rr]), __expf(-m_run[rr]));
    b_run[rr] = 1.f / (n + 1e-6f);
  }
#pragma unroll
  for (int dn = 0; dn < 8; dn++)
#pragma unroll
    for (int rr = 0; rr < 8; rr++) {
      const int s = s0 + rr + half * 8;
      hbuf[((size_t)(b * Sc + s) * NHc + h) * DHc + dn * 16 + r] = acc[dn][rr] * b_run[rr];
    }
}

// ---------------------------------------------------------------- LN + gate
// One wave per (b,s,head): per-head layernorm (weight only) * sigmoid(og),
// output bf16 [B*S, D].
__global__ void __launch_bounds__(256) ln_gate_kernel(const float* __restrict__ hbuf,
                                                      const float* __restrict__ og,
                                                      const float* __restrict__ lnw,
                                                      __bf16* __restrict__ hout) {
  const int wid  = blockIdx.x * 8 + (threadIdx.x >> 5);  // 0..B*S*NH-1
  const int lane = threadIdx.x & 31;
  const int hh   = wid & 7;
  const int bs   = wid >> 3;  // b*S+s
  const float* hp = hbuf + ((size_t)bs * NHc + hh) * DHc + lane * 4;
  float v[4];
#pragma unroll
  for (int j = 0; j < 4; j++) v[j] = hp[j];
  float sum = v[0] + v[1] + v[2] + v[3];
#pragma unroll
  for (int m = 1; m < 32; m <<= 1) sum += __shfl_xor(sum, m, 32);
  const float mu = sum * (1.f / DHc);
  float vs = 0.f;
#pragma unroll
  for (int j = 0; j < 4; j++) {
    float t = v[j] - mu;
    vs += t * t;
  }
#pragma unroll
  for (int m = 1; m < 32; m <<= 1) vs += __shfl_xor(vs, m, 32);
  const float rinv = rsqrtf(vs * (1.f / DHc) + 1e-6f);
#pragma unroll
  for (int j = 0; j < 4; j++) {
    const int d   = lane * 4 + j;
    float val     = (v[j] - mu) * rinv * lnw[hh * DHc + d];
    const float o = og[(size_t)bs * Dc + hh * DHc + d];
    val *= 1.f / (1.f + __expf(-o));
    hout[(size_t)bs * Dc + hh * DHc + d] = (__bf16)val;
  }
}

// ---------------------------------------------------------------- launch
extern "C" void kernel_launch(void* const* d_in, const int* in_sizes, int n_in,
                              void* d_out, int out_size, void* d_ws, size_t ws_size,
                              hipStream_t stream) {
  const float* x    = (const float*)d_in[0];
  const float* Wq   = (const float*)d_in[1];
  const float* Wk   = (const float*)d_in[2];
  const float* Wv   = (const float*)d_in[3];
  const float* Wog  = (const float*)d_in[4];
  const float* Wi   = (const float*)d_in[5];
  const float* biv  = (const float*)d_in[6];
  const float* Wf   = (const float*)d_in[7];
  const float* bfv  = (const float*)d_in[8];
  const float* lnw  = (const float*)d_in[9];
  const float* Wout = (const float*)d_in[10];
  float* out = (float*)d_out;

  const int MT = Bc * Sc;  // 4096 rows
  char* ws   = (char*)d_ws;
  size_t off = 0;
  auto alloc = [&](size_t bytes) -> char* {
    char* p = ws + off;
    off += (bytes + 255) & ~(size_t)255;
    return p;
  };
  __bf16* x_bf    = (__bf16*)alloc((size_t)MT * Dc * 2);
  __bf16* Wq_bf   = (__bf16*)alloc((size_t)Dc * Dc * 2);
  __bf16* Wk_bf   = (__bf16*)alloc((size_t)DHc * Dc * 2);
  __bf16* Wv_bf   = (__bf16*)alloc((size_t)DHc * Dc * 2);
  __bf16* Wog_bf  = (__bf16*)alloc((size_t)Dc * Dc * 2);
  __bf16* Wout_bf = (__bf16*)alloc((size_t)Dc * Dc * 2);
  float*  q_f32   = (float*)alloc((size_t)MT * Dc * 4);  // reused as h_buf
  __bf16* q_bf    = (__bf16*)alloc((size_t)MT * Dc * 2);
  float*  k_f32   = (float*)alloc((size_t)MT * DHc * 4);
  __bf16* k_bf    = (__bf16*)alloc((size_t)MT * DHc * 2);
  float*  v_f32   = (float*)alloc((size_t)MT * DHc * 4);
  __bf16* vT_bf   = (__bf16*)alloc((size_t)MT * DHc * 2);
  float*  og_f32  = (float*)alloc((size_t)MT * Dc * 4);
  float*  ipre    = (float*)alloc((size_t)Bc * NHc * Sc * 4);
  float*  logfv   = (float*)alloc((size_t)Bc * NHc * Sc * 4);
  float*  csum    = (float*)alloc((size_t)Bc * NHc * Sc * 4);
  float*  ucomb   = (float*)alloc((size_t)Bc * NHc * Sc * 4);
  __bf16* hout_bf = (__bf16*)alloc((size_t)MT * Dc * 2);
  float*  h_buf   = q_f32;  // q_f32 dead after q_bf conversion

  auto cvt = [&](const float* s, __bf16* d, int n) {
    cvt_bf16_kernel<<<(n + 255) / 256, 256, 0, stream>>>(s, d, n);
  };
  // 1. bf16 conversions of inputs/weights
  cvt(x, x_bf, MT * Dc);
  cvt(Wq, Wq_bf, Dc * Dc);
  cvt(Wk, Wk_bf, DHc * Dc);
  cvt(Wv, Wv_bf, DHc * Dc);
  cvt(Wog, Wog_bf, Dc * Dc);
  cvt(Wout, Wout_bf, Dc * Dc);
  // 2. projections (block tile 128x256; wave tile 64x64)
  dim3 blk(256);
  gemm_bf16_kernel<<<dim3(MT / 128, Dc / 256), blk, 0, stream>>>(x_bf, Wq_bf, q_f32, MT, Dc, Dc);
  gemm_bf16_kernel<<<dim3(MT / 128, 1), blk, 0, stream>>>(x_bf, Wk_bf, k_f32, MT, DHc, Dc);
  gemm_bf16_kernel<<<dim3(MT / 128, 1), blk, 0, stream>>>(x_bf, Wv_bf, v_f32, MT, DHc, Dc);
  gemm_bf16_kernel<<<dim3(MT / 128, Dc / 256), blk, 0, stream>>>(x_bf, Wog_bf, og_f32, MT, Dc, Dc);
  // 3. gates + cumulative log-forget
  gates_kernel<<<MT / 8, blk, 0, stream>>>(x, Wi, biv, Wf, bfv, ipre, logfv);
  scan_kernel<<<Bc * NHc, blk, 0, stream>>>(logfv, ipre, csum, ucomb);
  // 4. operand prep for the mLSTM core (fold DH^-0.5 into q)
  cvt_bf16_scale_kernel<<<(MT * Dc + 255) / 256, 256, 0, stream>>>(
      q_f32, q_bf, 0.08838834764831845f, MT * Dc);
  cvt(k_f32, k_bf, MT * DHc);
  transpose_v_kernel<<<(MT * DHc + 255) / 256, 256, 0, stream>>>(v_f32, vT_bf);
  // 5. flash-style stabilized parallel mLSTM
  mlstm_attn_kernel<<<(Bc * NHc * (Sc / 16)) / 8, blk, 0, stream>>>(q_bf, k_bf, vT_bf, csum,
                                                                    ucomb, h_buf);
  // 6. per-head layernorm + output gate
  ln_gate_kernel<<<(MT * NHc) / 8, blk, 0, stream>>>(h_buf, og_f32, lnw, hout_bf);
  // 7. output projection
  gemm_bf16_kernel<<<dim3(MT / 128, Dc / 256), blk, 0, stream>>>(hout_bf, Wout_bf, out, MT, Dc, Dc);
}
